// WaveNetGlow_24876450578987
// MI455X (gfx1250) — matrix operand order repaired and checked by reading, not compiled, not run on hardware
//
#include <hip/hip_runtime.h>
#include <hip/hip_bf16.h>

// WaveNet/WaveGlow WN stack for MI455X (gfx1250), bf16 WMMA formulation.
// Dilated convs / res-skip / cond 1x1 all run on v_wmma_f32_16x16x32_bf16.
// B panels are staged cooperatively into LDS (async-to-LDS when available,
// double-buffered) to cut the 4x redundant L2 reads; fp32 accumulation.

#define TQ    4096
#define BGN   4
#define NCHN  256
#define NLAY  8
#define HPAD  128
#define HROWS (TQ + 2*HPAD)   // 4352

typedef __attribute__((ext_vector_type(16))) __bf16 v16bf;
typedef __attribute__((ext_vector_type(8)))  float  v8f;
typedef int v4i_ __attribute__((vector_size(16)));   // matches builtin param type

union Frag32 { uint4 q[2]; v16bf v; };

#if __has_builtin(__builtin_amdgcn_global_load_async_to_lds_b128) && \
    __has_builtin(__builtin_amdgcn_s_wait_asynccnt)
#define USE_ASYNC_LDS 1
#else
#define USE_ASYNC_LDS 0
#endif

__device__ inline void wait_async0() {
#if USE_ASYNC_LDS
  __builtin_amdgcn_s_wait_asynccnt(0);
#endif
}

// copy 16B global -> LDS (async DMA when available; ASYNCcnt-tracked)
__device__ inline void stage16(const unsigned short* __restrict__ g,
                               unsigned short* __restrict__ l) {
#if USE_ASYNC_LDS
  __builtin_amdgcn_global_load_async_to_lds_b128(
      (__attribute__((address_space(1))) v4i_*)g,
      (__attribute__((address_space(3))) v4i_*)l, 0, 0);
#else
  *(uint4*)l = *(const uint4*)g;
#endif
}

__device__ inline v8f wmma_bf16(v16bf a, v16bf b, v8f c) {
  // D = A(16x32) * B(32x16) + C(16x16 f32)
  return __builtin_amdgcn_wmma_f32_16x16x32_bf16(false, a, false, b, (short)0, c, false, false);
}

__device__ inline unsigned short f2bf(float f) {
  unsigned u = __float_as_uint(f);
  u += 0x7fffu + ((u >> 16) & 1u);   // round-to-nearest-even
  return (unsigned short)(u >> 16);
}
__device__ inline float bf2f(unsigned short b) {
  return __uint_as_float(((unsigned)b) << 16);
}

// A-matrix 16x32 K-mapping (contiguous, matches B fills):
// lanes 0-15 : V0..V7 -> K 0..15
// lanes 16-31: V0..V7 -> K 16..31
__device__ inline int a_kmap(int lane, int i) {
  return (lane >> 4) * 16 + i;
}

// ---------------- packing / prep kernels ----------------

__global__ void k_pack2d(const float* __restrict__ src, unsigned short* __restrict__ dst,
                         int C, int Cpad, int total) {
  int idx = blockIdx.x * 256 + threadIdx.x;
  if (idx >= total) return;
  int i = idx & 15, lane = (idx >> 4) & 31, tile = idx >> 9;
  int nKc = Cpad >> 5;
  int rowtile = tile / nKc, kc = tile - rowtile * nKc;
  int row = rowtile * 16 + (lane & 15);
  int col = kc * 32 + a_kmap(lane, i);
  float v = (col < C) ? src[(size_t)row * C + col] : 0.f;
  dst[idx] = f2bf(v);
}

// Pack in_w [8][512][256][3]: tile = ((layer*3+k)*32 + rowtile)*8 + kc
__global__ void k_pack_inw(const float* __restrict__ src, unsigned short* __restrict__ dst,
                           int total) {
  int idx = blockIdx.x * 256 + threadIdx.x;
  if (idx >= total) return;
  int i = idx & 15, lane = (idx >> 4) & 31, tile = idx >> 9;
  int layer = tile / 768, rem = tile % 768;
  int k = rem >> 8, rowtile = (rem >> 3) & 31, kc = rem & 7;
  int o = rowtile * 16 + (lane & 15);
  int c = kc * 32 + a_kmap(lane, i);
  dst[idx] = f2bf(src[((size_t)(layer * 512 + o) * 256 + c) * 3 + k]);
}

// spect [80][4096] f32 -> spectT [4096][96] bf16 (K padded to 96)
__global__ void k_spectT(const float* __restrict__ spect, unsigned short* __restrict__ dst) {
  int idx = blockIdx.x * 256 + threadIdx.x;
  if (idx >= TQ * 96) return;
  int t = idx / 96, c = idx - t * 96;
  float v = (c < 80) ? spect[(size_t)c * TQ + t] : 0.f;
  dst[idx] = f2bf(v);
}

// start conv: h = start_w[256,8] @ audio + start_b -> hf [bg][t][256] f32, hT bf16 (halo'd)
__global__ void k_start(const float* __restrict__ audio, const float* __restrict__ sw,
                        const float* __restrict__ sb, float* __restrict__ hf,
                        unsigned short* __restrict__ hT) {
  int idx = blockIdx.x * 256 + threadIdx.x;           // 4*4096*256 = 2^22
  int o = idx & 255, t = (idx >> 8) & 4095, bg = idx >> 20;
  float s = sb[o];
#pragma unroll
  for (int c = 0; c < 8; ++c)
    s += sw[o * 8 + c] * audio[((size_t)(bg * 8 + c)) * TQ + t];
  hf[idx] = s;                                         // idx == (bg*TQ+t)*256+o
  hT[((size_t)(bg * HROWS + HPAD + t)) * NCHN + o] = f2bf(s);
}

// pcb[bg][ch] = relu(pos_emb_w[pos[bg]]) @ pos_lin_w[ch].T + pos_lin_b[ch] + in_b[ch]
__global__ void k_pcb(const int* __restrict__ pos, const float* __restrict__ pew,
                      const float* __restrict__ plw, const float* __restrict__ plb,
                      const float* __restrict__ inb, float* __restrict__ pcb) {
  int idx = blockIdx.x * 256 + threadIdx.x;            // 4*4096
  if (idx >= BGN * 4096) return;
  int bg = idx >> 12, ch = idx & 4095;
  const float* pe = pew + pos[bg] * 128;
  const float* w  = plw + (size_t)ch * 128;
  float s = plb[ch] + inb[ch];
#pragma unroll 8
  for (int j = 0; j < 128; ++j) s += fmaxf(pe[j], 0.f) * w[j];
  pcb[idx] = s;
}

// ---------------- WMMA GEMM kernels ----------------

// cond = cond_w[4096,80] @ spect[80,4096] + cond_b  -> condb [4096 ch][4096 t] bf16
__global__ __launch_bounds__(256) void k_cond(const unsigned short* __restrict__ spectT,
                                              const unsigned short* __restrict__ wpack,
                                              const float* __restrict__ cond_b,
                                              unsigned short* __restrict__ condb) {
  int t0 = blockIdx.x * 128, ch0 = blockIdx.y * 64;
  int wave = threadIdx.x >> 5, lane = threadIdx.x & 31;
  int r = wave & 3, cg = (wave >> 2) * 4;
  int lm = lane & 15, lh = lane >> 4;
  int rt = (ch0 >> 4) + r;
  v8f zero = {};
  v8f acc[4] = {zero, zero, zero, zero};
#pragma unroll
  for (int kc = 0; kc < 3; ++kc) {
    Frag32 a;
    const uint4* pa = (const uint4*)(wpack + (((size_t)(rt * 3 + kc)) << 9) + (size_t)lane * 16);
    a.q[0] = pa[0]; a.q[1] = pa[1];
    Frag32 bfr[4];
    int ch = kc * 32 + lh * 16;
#pragma unroll
    for (int c = 0; c < 4; ++c) {
      const uint4* pb = (const uint4*)(spectT + (size_t)(t0 + 16 * (cg + c) + lm) * 96 + ch);
      bfr[c].q[0] = pb[0]; bfr[c].q[1] = pb[1];
    }
#pragma unroll
    for (int c = 0; c < 4; ++c) acc[c] = wmma_bf16(a.v, bfr[c].v, acc[c]);
  }
#pragma unroll
  for (int c = 0; c < 4; ++c) {
    int t = t0 + (cg + c) * 16 + lm;
#pragma unroll
    for (int v = 0; v < 8; ++v) {
      int ch = ch0 + r * 16 + v + lh * 8;
      condb[(size_t)ch * TQ + t] = f2bf(acc[c][v] + cond_b[ch]);
    }
  }
}

// layer gate: in_act = sum_k W_k @ shift(h,(k-1)d) + cond + pcb ; acts = tanh*sigmoid.
// B panels (32ch x 128t bf16, 8KB) staged into double-buffered LDS via async DMA.
__global__ __launch_bounds__(256) void k_layer_gate(const unsigned short* __restrict__ hT,
                                                    const unsigned short* __restrict__ wpack,
                                                    const unsigned short* __restrict__ condb,
                                                    const float* __restrict__ pcb,
                                                    unsigned short* __restrict__ actsT,
                                                    int layer, int dil) {
  __shared__ __align__(16) unsigned short sB[2][128 * 32];
  __shared__ __align__(16) unsigned short sAct[128 * 64];
  int bg = blockIdx.z, t0 = blockIdx.x * 128, gc0 = blockIdx.y * 64;
  int wave = threadIdx.x >> 5, lane = threadIdx.x & 31;
  int r = wave & 3, cg = (wave >> 2) * 4;
  int lm = lane & 15, lh = lane >> 4;
  int rt_t = (gc0 >> 4) + r;        // tanh rows tile; sigmoid tile = rt_t + 16
  v8f zero = {};
  v8f acc[2][4];
#pragma unroll
  for (int h = 0; h < 2; ++h)
#pragma unroll
    for (int c = 0; c < 4; ++c) acc[h][c] = zero;

  const unsigned short* hb = hT + ((size_t)bg * HROWS + HPAD + t0) * NCHN;
  // thread's two 16B staging chunks: chunk = t_local*4 + quarter
  int lin0 = threadIdx.x * 2;

  auto stage = [&](int s) {
    int k = s >> 3, kc = s & 7;
    const unsigned short* hk = hb + (ptrdiff_t)(k - 1) * dil * NCHN + kc * 32;
    unsigned short* dst = sB[s & 1];
#pragma unroll
    for (int j = 0; j < 2; ++j) {
      int tl = (lin0 + j) >> 2, q = (lin0 + j) & 3;
      stage16(hk + (size_t)tl * NCHN + q * 8, dst + tl * 32 + q * 8);
    }
  };

  stage(0);
#pragma unroll 2
  for (int s = 0; s < 24; ++s) {
    wait_async0();          // own stage(s) complete
    __syncthreads();        // all stage(s) visible; all compute(s-1) done
    if (s + 1 < 24) stage(s + 1);
    int k = s >> 3, kc = s & 7;
    int ta = ((layer * 3 + k) * 32 + rt_t) * 8 + kc;
    Frag32 a0, a1;
    const uint4* pa0 = (const uint4*)(wpack + (((size_t)ta) << 9) + (size_t)lane * 16);
    const uint4* pa1 = (const uint4*)(wpack + (((size_t)(ta + 128)) << 9) + (size_t)lane * 16);
    a0.q[0] = pa0[0]; a0.q[1] = pa0[1];
    a1.q[0] = pa1[0]; a1.q[1] = pa1[1];
    const unsigned short* buf = sB[s & 1];
#pragma unroll
    for (int c = 0; c < 4; ++c) {
      const uint4* pb = (const uint4*)(buf + (16 * (cg + c) + lm) * 32 + lh * 16);
      Frag32 b; b.q[0] = pb[0]; b.q[1] = pb[1];
      acc[0][c] = wmma_bf16(a0.v, b.v, acc[0][c]);
      acc[1][c] = wmma_bf16(a1.v, b.v, acc[1][c]);
    }
  }
  __syncthreads();
  int chT0 = layer * 512;
#pragma unroll
  for (int c = 0; c < 4; ++c) {
    int t = t0 + (cg + c) * 16 + lm;
#pragma unroll
    for (int v = 0; v < 8; ++v) {
      int gc = gc0 + r * 16 + v + lh * 8;
      float bt = bf2f(condb[(size_t)(chT0 + gc) * TQ + t]) + pcb[bg * 4096 + chT0 + gc];
      float bs = bf2f(condb[(size_t)(chT0 + 256 + gc) * TQ + t]) + pcb[bg * 4096 + chT0 + 256 + gc];
      float xt = acc[0][c][v] + bt;
      float xs = acc[1][c][v] + bs;
      float g = tanhf(xt) * (1.0f / (1.0f + __expf(-xs)));
      sAct[((cg + c) * 16 + lm) * 64 + (r * 16 + v + lh * 8)] = f2bf(g);
    }
  }
  __syncthreads();
#pragma unroll
  for (int j = 0; j < 4; ++j) {
    int idx4 = threadIdx.x + j * 256;
    int tl = idx4 >> 3, q = idx4 & 7;
    *(uint4*)(actsT + ((size_t)(bg * TQ + t0 + tl)) * NCHN + gc0 + q * 8) =
        *(const uint4*)(sAct + tl * 64 + q * 8);
  }
}

// res/skip: rs = rs_w @ acts (+b). h += rs[:256] (-> fp32 + bf16 hT); out += rs[256:].
// last==1: rs_last_w [256,256], only out accumulation. LDS-staged B panels.
__global__ __launch_bounds__(256) void k_rs(const unsigned short* __restrict__ actsT,
                                            const unsigned short* __restrict__ wpack,
                                            const float* __restrict__ rsb,
                                            float* __restrict__ hf,
                                            unsigned short* __restrict__ hT,
                                            float* __restrict__ outT, int last) {
  __shared__ __align__(16) unsigned short sB[2][128 * 32];
  __shared__ __align__(16) float sH[128 * 64];
  __shared__ __align__(16) float sO[128 * 64];
  int bg = blockIdx.z, t0 = blockIdx.x * 128, gc0 = blockIdx.y * 64;
  int wave = threadIdx.x >> 5, lane = threadIdx.x & 31;
  int r = wave & 3, cg = (wave >> 2) * 4;
  int lm = lane & 15, lh = lane >> 4;
  int rt = (gc0 >> 4) + r;
  v8f zero = {};
  v8f acc0[4] = {zero, zero, zero, zero};
  v8f acc1[4] = {zero, zero, zero, zero};
  const unsigned short* ab = actsT + (size_t)(bg * TQ + t0) * NCHN;
  int lin0 = threadIdx.x * 2;

  auto stage = [&](int s) {
    const unsigned short* src = ab + s * 32;
    unsigned short* dst = sB[s & 1];
#pragma unroll
    for (int j = 0; j < 2; ++j) {
      int tl = (lin0 + j) >> 2, q = (lin0 + j) & 3;
      stage16(src + (size_t)tl * NCHN + q * 8, dst + tl * 32 + q * 8);
    }
  };

  stage(0);
#pragma unroll 2
  for (int kc = 0; kc < 8; ++kc) {
    wait_async0();
    __syncthreads();
    if (kc + 1 < 8) stage(kc + 1);
    int ta = rt * 8 + kc;
    Frag32 a0, a1;
    const uint4* pa0 = (const uint4*)(wpack + (((size_t)ta) << 9) + (size_t)lane * 16);
    a0.q[0] = pa0[0]; a0.q[1] = pa0[1];
    if (!last) {
      const uint4* pa1 = (const uint4*)(wpack + (((size_t)(ta + 128)) << 9) + (size_t)lane * 16);
      a1.q[0] = pa1[0]; a1.q[1] = pa1[1];
    }
    const unsigned short* buf = sB[kc & 1];
#pragma unroll
    for (int c = 0; c < 4; ++c) {
      const uint4* pb = (const uint4*)(buf + (16 * (cg + c) + lm) * 32 + lh * 16);
      Frag32 b; b.q[0] = pb[0]; b.q[1] = pb[1];
      acc0[c] = wmma_bf16(a0.v, b.v, acc0[c]);
      if (!last) acc1[c] = wmma_bf16(a1.v, b.v, acc1[c]);
    }
  }
  __syncthreads();
#pragma unroll
  for (int c = 0; c < 4; ++c) {
    int tl = (cg + c) * 16 + lm;
#pragma unroll
    for (int v = 0; v < 8; ++v) {
      int cl = r * 16 + v + lh * 8;        // channel within 64-group
      int o = gc0 + cl;
      if (!last) {
        sH[tl * 64 + cl] = acc0[c][v] + rsb[o];
        sO[tl * 64 + cl] = acc1[c][v] + rsb[256 + o];
      } else {
        sO[tl * 64 + cl] = acc0[c][v] + rsb[o];
      }
    }
  }
  __syncthreads();
#pragma unroll
  for (int j = 0; j < 8; ++j) {
    int idx4 = threadIdx.x + j * 256;
    int tl = idx4 >> 4, q = idx4 & 15;
    size_t base = (size_t)(bg * TQ + t0 + tl) * NCHN + gc0 + q * 4;
    if (!last) {
      float4 hv = *(float4*)(hf + base);
      float4 dh = *(const float4*)(sH + tl * 64 + q * 4);
      hv.x += dh.x; hv.y += dh.y; hv.z += dh.z; hv.w += dh.w;
      *(float4*)(hf + base) = hv;
      ushort4 hb4;
      hb4.x = f2bf(hv.x); hb4.y = f2bf(hv.y); hb4.z = f2bf(hv.z); hb4.w = f2bf(hv.w);
      *(ushort4*)(hT + (size_t)(bg * HROWS + HPAD + t0 + tl) * NCHN + gc0 + q * 4) = hb4;
    }
    float4 ov = *(float4*)(outT + base);
    float4 dv = *(const float4*)(sO + tl * 64 + q * 4);
    ov.x += dv.x; ov.y += dv.y; ov.z += dv.z; ov.w += dv.w;
    *(float4*)(outT + base) = ov;
  }
}

// end conv: y = end_w[16,256] @ out + end_b; split halves into d_out
__global__ void k_end(const float* __restrict__ outT, const float* __restrict__ ew,
                      const float* __restrict__ eb, float* __restrict__ y) {
  int idx = blockIdx.x * 256 + threadIdx.x;            // 4*4096
  if (idx >= BGN * TQ) return;
  int bg = idx >> 12, t = idx & 4095;
  const float* row = outT + (size_t)idx * NCHN;
  float acc[16];
#pragma unroll
  for (int oc = 0; oc < 16; ++oc) acc[oc] = eb[oc];
  for (int c = 0; c < NCHN; c += 4) {
    float4 x = *(const float4*)(row + c);
#pragma unroll
    for (int oc = 0; oc < 16; ++oc) {
      const float* w = ew + oc * NCHN + c;
      acc[oc] += w[0] * x.x + w[1] * x.y + w[2] * x.z + w[3] * x.w;
    }
  }
#pragma unroll
  for (int oc = 0; oc < 8; ++oc)
    y[(size_t)(bg * 8 + oc) * TQ + t] = acc[oc];
#pragma unroll
  for (int oc = 8; oc < 16; ++oc)
    y[(size_t)BGN * 8 * TQ + (size_t)(bg * 8 + oc - 8) * TQ + t] = acc[oc];
}

// ---------------- host launcher ----------------

extern "C" void kernel_launch(void* const* d_in, const int* in_sizes, int n_in,
                              void* d_out, int out_size, void* d_ws, size_t ws_size,
                              hipStream_t stream) {
  const float* audio     = (const float*)d_in[0];
  const float* spect     = (const float*)d_in[1];
  const int*   pos       = (const int*)d_in[2];
  const float* start_w   = (const float*)d_in[3];
  const float* start_b   = (const float*)d_in[4];
  const float* cond_w    = (const float*)d_in[5];
  const float* cond_b    = (const float*)d_in[6];
  const float* pos_emb_w = (const float*)d_in[7];
  const float* pos_lin_w = (const float*)d_in[8];
  const float* pos_lin_b = (const float*)d_in[9];
  const float* in_w      = (const float*)d_in[10];
  const float* in_b      = (const float*)d_in[11];
  const float* rs_w      = (const float*)d_in[12];
  const float* rs_b      = (const float*)d_in[13];
  const float* rs_last_w = (const float*)d_in[14];
  const float* rs_last_b = (const float*)d_in[15];
  const float* end_w     = (const float*)d_in[16];
  const float* end_b     = (const float*)d_in[17];

  char* ws = (char*)d_ws;
  size_t off = 0;
  auto alloc = [&](size_t bytes) { size_t o = off; off += (bytes + 255) & ~(size_t)255; return o; };
  unsigned short* hT     = (unsigned short*)(ws + alloc((size_t)BGN * HROWS * NCHN * 2));
  float*          hf     = (float*)(ws + alloc((size_t)BGN * TQ * NCHN * 4));
  unsigned short* actsT  = (unsigned short*)(ws + alloc((size_t)BGN * TQ * NCHN * 2));
  float*          outT   = (float*)(ws + alloc((size_t)BGN * TQ * NCHN * 4));
  unsigned short* condbf = (unsigned short*)(ws + alloc((size_t)4096 * TQ * 2));
  unsigned short* spectT = (unsigned short*)(ws + alloc((size_t)TQ * 96 * 2));
  float*          pcb    = (float*)(ws + alloc((size_t)BGN * 4096 * 4));
  unsigned short* pIn    = (unsigned short*)(ws + alloc((size_t)8 * 3 * 32 * 8 * 512 * 2));
  unsigned short* pRs    = (unsigned short*)(ws + alloc((size_t)7 * 32 * 8 * 512 * 2));
  unsigned short* pRsL   = (unsigned short*)(ws + alloc((size_t)16 * 8 * 512 * 2));
  unsigned short* pCw    = (unsigned short*)(ws + alloc((size_t)256 * 3 * 512 * 2));

  (void)hipMemsetAsync(hT, 0, (size_t)BGN * HROWS * NCHN * 2, stream);  // zero halo (+interior)
  (void)hipMemsetAsync(outT, 0, (size_t)BGN * TQ * NCHN * 4, stream);

  int totIn = 8 * 3 * 32 * 8 * 512;        // 3,145,728
  k_pack_inw<<<(totIn + 255) / 256, 256, 0, stream>>>(in_w, pIn, totIn);
  int totRs = 7 * 32 * 8 * 512;
  k_pack2d<<<(totRs + 255) / 256, 256, 0, stream>>>(rs_w, pRs, 256, 256, totRs);
  int totRsL = 16 * 8 * 512;
  k_pack2d<<<(totRsL + 255) / 256, 256, 0, stream>>>(rs_last_w, pRsL, 256, 256, totRsL);
  int totCw = 256 * 3 * 512;
  k_pack2d<<<(totCw + 255) / 256, 256, 0, stream>>>(cond_w, pCw, 80, 96, totCw);
  k_spectT<<<(TQ * 96 + 255) / 256, 256, 0, stream>>>(spect, spectT);
  k_start<<<(BGN * TQ * NCHN) / 256, 256, 0, stream>>>(audio, start_w, start_b, hf, hT);
  k_pcb<<<(BGN * 4096 + 255) / 256, 256, 0, stream>>>(pos, pos_emb_w, pos_lin_w,
                                                      pos_lin_b, in_b, pcb);
  k_cond<<<dim3(TQ / 128, 4096 / 64, 1), 256, 0, stream>>>(spectT, pCw, cond_b, condbf);

  for (int i = 0; i < NLAY; ++i) {
    k_layer_gate<<<dim3(TQ / 128, 4, BGN), 256, 0, stream>>>(hT, pIn, condbf, pcb,
                                                             actsT, i, 1 << i);
    if (i < NLAY - 1) {
      k_rs<<<dim3(TQ / 128, 4, BGN), 256, 0, stream>>>(actsT, pRs + (size_t)i * 32 * 8 * 512,
                                                       rs_b + i * 512, hf, hT, outT, 0);
    } else {
      k_rs<<<dim3(TQ / 128, 4, BGN), 256, 0, stream>>>(actsT, pRsL, rs_last_b,
                                                       hf, hT, outT, 1);
    }
  }
  k_end<<<(BGN * TQ + 255) / 256, 256, 0, stream>>>(outT, end_w, end_b, (float*)d_out);
}
